// GATLayer_84301618086306
// MI455X (gfx1250) — matrix operand order repaired and checked by reading, not compiled, not run on hardware
//
#include <hip/hip_runtime.h>
#include <math.h>

// MI455X / gfx1250, wave32. WMMA bf16 16x16x32, f32 accumulate.
typedef __attribute__((ext_vector_type(16))) __bf16 v16bf;
typedef __attribute__((ext_vector_type(8)))  float  v8f;

#define GB 8
#define GN 2048
#define GFI 128
#define GFO 64
#define GFP 80   // 64 out cols + 1 denominator col, padded to 5 N-tiles

__device__ __forceinline__ __bf16 f2bf(float x) { return (__bf16)x; }

__device__ __forceinline__ void splitbf(float x, __bf16& hi, __bf16& lo) {
    hi = (__bf16)x;
    lo = (__bf16)(x - (float)hi);
}

// -------------------------------------------------------------------------
// K1: h = x @ W  (per-batch 2048x128x64) with bf16 hi/lo split WMMA
//     (3 WMMAs per tile -> ~f32 accuracy), plus s2[b,j] = h[b,j,:] . a2
// 1 wave per 16-row M-tile, all 4 N-tiles. grid = (8*128)/8 blocks of 256.
// h is written into d_out (reused as scratch; overwritten by K4 later).
// -------------------------------------------------------------------------
__global__ __launch_bounds__(256) void gat_hgemm(
        const float* __restrict__ x, const float* __restrict__ W,
        const float* __restrict__ a, float* __restrict__ h,
        float* __restrict__ s2) {
    const int tid  = threadIdx.x;
    const int gw   = blockIdx.x * 8 + (tid >> 5);   // global wave id
    const int b    = gw >> 7;                        // 128 M-tiles per batch
    const int mt   = gw & 127;
    const int mi   = mt * 16;
    const int L    = tid & 31;
    const int m    = L & 15;
    const int half = L >> 4;

    const float* xrow = x + ((size_t)b * GN + mi + m) * GFI;
    const float* a2   = a + GFO;

    v8f acc[4];
    #pragma unroll
    for (int i = 0; i < 4; ++i) acc[i] = (v8f){0,0,0,0,0,0,0,0};

    #pragma unroll
    for (int ks = 0; ks < 4; ++ks) {
        const int kb = ks * 32;
        // ---- A operand: 16x32 f32 rows of x -> hi/lo bf16 (documented A layout)
        float af[16];
        {
            float4 c0 = *(const float4*)(xrow + kb + half * 8);
            float4 c1 = *(const float4*)(xrow + kb + half * 8 + 4);
            float4 c2 = *(const float4*)(xrow + kb + 16 + half * 8);
            float4 c3 = *(const float4*)(xrow + kb + 16 + half * 8 + 4);
            af[0]=c0.x; af[1]=c0.y; af[2]=c0.z; af[3]=c0.w;
            af[4]=c1.x; af[5]=c1.y; af[6]=c1.z; af[7]=c1.w;
            af[8]=c2.x; af[9]=c2.y; af[10]=c2.z; af[11]=c2.w;
            af[12]=c3.x; af[13]=c3.y; af[14]=c3.z; af[15]=c3.w;
        }
        v16bf Ahi, Alo;
        #pragma unroll
        for (int i = 0; i < 16; ++i) { __bf16 hi, lo; splitbf(af[i], hi, lo); Ahi[i] = hi; Alo[i] = lo; }

        #pragma unroll
        for (int nt = 0; nt < 4; ++nt) {
            const int n = nt * 16 + m;
            v16bf Bhi, Blo;
            #pragma unroll
            for (int v = 0; v < 16; ++v) {
                float wv = W[(kb + ((v < 8) ? (half * 8 + v) : (16 + half * 8 + v - 8))) * GFO + n];
                __bf16 hi, lo; splitbf(wv, hi, lo);
                Bhi[v] = hi; Blo[v] = lo;
            }
            acc[nt] = __builtin_amdgcn_wmma_f32_16x16x32_bf16(false, Ahi, false, Bhi, (short)0, acc[nt], false, false);
            acc[nt] = __builtin_amdgcn_wmma_f32_16x16x32_bf16(false, Ahi, false, Blo, (short)0, acc[nt], false, false);
            acc[nt] = __builtin_amdgcn_wmma_f32_16x16x32_bf16(false, Alo, false, Bhi, (short)0, acc[nt], false, false);
        }
    }

    // ---- store h + reduce s2 = h . a2
    // C layout: lane L, vgpr v -> row = (L>>4)*8 + v, col = nt*16 + (L&15)
    #pragma unroll
    for (int v = 0; v < 8; ++v) {
        const int row = mi + half * 8 + v;
        float part = 0.0f;
        #pragma unroll
        for (int nt = 0; nt < 4; ++nt) {
            const int col = nt * 16 + m;
            h[((size_t)b * GN + row) * GFO + col] = acc[nt][v];
            part += acc[nt][v] * a2[col];
        }
        // reduce across the 16 lanes of this half (xor stays inside the group)
        part += __shfl_xor(part, 1);
        part += __shfl_xor(part, 2);
        part += __shfl_xor(part, 4);
        part += __shfl_xor(part, 8);
        if (m == 0) s2[(size_t)b * GN + row] = part;
    }
}

// -------------------------------------------------------------------------
// K2: per-batch global max of s2 (for a numerically safe shared exp shift)
// -------------------------------------------------------------------------
__global__ __launch_bounds__(256) void gat_rowmax(
        const float* __restrict__ s2, float* __restrict__ mmax) {
    __shared__ float sm[256];
    const int b = blockIdx.x, tid = threadIdx.x;
    float v = -3.0e38f;
    for (int i = tid; i < GN; i += 256) v = fmaxf(v, s2[(size_t)b * GN + i]);
    sm[tid] = v;
    __syncthreads();
    for (int s = 128; s > 0; s >>= 1) {
        if (tid < s) sm[tid] = fmaxf(sm[tid], sm[tid + s]);
        __syncthreads();
    }
    if (tid == 0) mmax[b] = sm[0];
}

// -------------------------------------------------------------------------
// K3: build transposed G_t[b][f][j] (bf16):
//     f<64: p[j]*h[j,f];  f==64: p[j];  f in 65..79: 0  (p = exp(s2 - max))
// Transposed layout makes K4's B-operand a contiguous 32B load per lane.
// grid = 8 batches * 16 j-tiles of 128, 256 threads.
// -------------------------------------------------------------------------
__global__ __launch_bounds__(256) void gat_buildg(
        const float* __restrict__ h, const float* __restrict__ s2,
        const float* __restrict__ mmax, unsigned short* __restrict__ gt_raw) {
    __shared__ float hl[128 * 65];   // +1 pad: conflict-free transpose reads
    __shared__ float pl[128];
    __bf16* gt = (__bf16*)gt_raw;

    const int tid = threadIdx.x;
    const int b   = blockIdx.x >> 4;
    const int j0  = (blockIdx.x & 15) * 128;

    const float* hb = h + ((size_t)b * GN + j0) * GFO;
    for (int idx = tid; idx < 128 * GFO; idx += 256) {
        const int j = idx >> 6, f = idx & 63;
        hl[j * 65 + f] = hb[idx];
    }
    if (tid < 128) pl[tid] = __expf(s2[(size_t)b * GN + j0 + tid] - mmax[b]);
    __syncthreads();

    for (int idx = tid; idx < 128 * GFP; idx += 256) {
        const int f = idx >> 7, j = idx & 127;
        float val = (f < GFO) ? pl[j] * hl[j * 65 + f]
                              : ((f == GFO) ? pl[j] : 0.0f);
        gt[((size_t)b * GFP + f) * GN + j0 + j] = f2bf(val);
    }
}

// -------------------------------------------------------------------------
// K4: out[i,0:64] = (mask @ G)[i,0:64] / (mask @ G)[i,64] + bias
// mask[i,j] = (adj[i,j] >= 1e-9 || i==j) ? 1 : 0  -- exact in bf16.
// 1 workgroup per (b, 16-row tile); 8 waves split K=2048 into chunks of 256;
// LDS reduction + fused divide/bias epilogue.
// -------------------------------------------------------------------------
__global__ __launch_bounds__(256) void gat_attn(
        const float* __restrict__ adj, const unsigned short* __restrict__ gt_raw,
        const float* __restrict__ bias, float* __restrict__ out) {
    __shared__ float red[8 * 32 * 40];   // 40 KB: 8 waves x 32 lanes x (5 tiles * 8)
    const __bf16* gt = (const __bf16*)gt_raw;

    const int tid  = threadIdx.x;
    const int w    = tid >> 5;
    const int L    = tid & 31;
    const int m    = L & 15;
    const int half = L >> 4;
    const int b    = blockIdx.x >> 7;
    const int mt   = blockIdx.x & 127;
    const int r    = mt * 16 + m;        // this lane's mask row

    const float*  adjrow = adj + ((size_t)b * GN + r) * GN;
    const __bf16* gtb    = gt + (size_t)b * GFP * GN;
    const __bf16  one    = (__bf16)1.0f;
    const __bf16  zero   = (__bf16)0.0f;

    v8f acc[5];
    #pragma unroll
    for (int i = 0; i < 5; ++i) acc[i] = (v8f){0,0,0,0,0,0,0,0};

    const int k0 = w * 256;
    for (int s = 0; s < 8; ++s) {
        const int kb = k0 + s * 32;
        // ---- A operand: 16x32 bf16 mask tile built from adj f32 (read once)
        const int ja = kb + half * 16;
        const int jb = kb + half * 16 + 8;
        float af[16];
        {
            float4 c0 = *(const float4*)(adjrow + ja);
            float4 c1 = *(const float4*)(adjrow + ja + 4);
            float4 c2 = *(const float4*)(adjrow + jb);
            float4 c3 = *(const float4*)(adjrow + jb + 4);
            af[0]=c0.x; af[1]=c0.y; af[2]=c0.z; af[3]=c0.w;
            af[4]=c1.x; af[5]=c1.y; af[6]=c1.z; af[7]=c1.w;
            af[8]=c2.x; af[9]=c2.y; af[10]=c2.z; af[11]=c2.w;
            af[12]=c3.x; af[13]=c3.y; af[14]=c3.z; af[15]=c3.w;
        }
        v16bf A;
        #pragma unroll
        for (int i = 0; i < 16; ++i) {
            const int j = (i < 8) ? (ja + i) : (jb + i - 8);
            A[i] = (af[i] >= 1e-9f || j == r) ? one : zero;
        }
        // ---- 5 N-tiles: cols 0..63 numerator + col 64 denominator
        #pragma unroll
        for (int nt = 0; nt < 5; ++nt) {
            const v16bf Bv = *(const v16bf*)(gtb + ((size_t)(nt * 16 + m)) * GN + kb + half * 16);
            acc[nt] = __builtin_amdgcn_wmma_f32_16x16x32_bf16(false, A, false, Bv, (short)0, acc[nt], false, false);
        }
    }

    // ---- cross-wave reduction in LDS
    const int base = (w * 32 + L) * 40;
    #pragma unroll
    for (int nt = 0; nt < 5; ++nt)
        #pragma unroll
        for (int v = 0; v < 8; ++v)
            red[base + nt * 8 + v] = acc[nt][v];
    __syncthreads();

    for (int slot = tid; slot < 1280; slot += 256) {
        float sum = 0.0f;
        #pragma unroll
        for (int w2 = 0; w2 < 8; ++w2) sum += red[w2 * 1280 + slot];
        red[slot] = sum;   // each slot owned by exactly one thread
    }
    __syncthreads();

    // ---- epilogue: divide numerator by denominator column, add bias
    for (int idx = tid; idx < 16 * GFO; idx += 256) {
        const int row = idx >> 6;
        const int col = idx & 63;
        const int Ln  = ((row >> 3) << 4) | (col & 15);
        const int Ld  = (row >> 3) << 4;            // denom col 64 -> lane&15 == 0
        const float num = red[Ln * 40 + (col >> 4) * 8 + (row & 7)];
        const float den = red[Ld * 40 + 4 * 8 + (row & 7)];
        out[((size_t)b * GN + mt * 16 + row) * GFO + col] = num / den + bias[col];
    }
}

// -------------------------------------------------------------------------
extern "C" void kernel_launch(void* const* d_in, const int* in_sizes, int n_in,
                              void* d_out, int out_size, void* d_ws, size_t ws_size,
                              hipStream_t stream) {
    const float* x    = (const float*)d_in[0];
    const float* adj  = (const float*)d_in[1];
    const float* W    = (const float*)d_in[2];
    const float* a    = (const float*)d_in[3];
    const float* bias = (const float*)d_in[4];

    float* out = (float*)d_out;
    float* h   = (float*)d_out;                       // staged h (overwritten by K4)

    char* ws = (char*)d_ws;
    float*          s2   = (float*)ws;                              // 8*2048*4   = 64 KB
    float*          mmax = (float*)(ws + GB * GN * 4);              // 32 B
    unsigned short* gt   = (unsigned short*)(ws + GB * GN * 4 + 256); // 8*80*2048*2 = 2.5 MB

    gat_hgemm <<< (GB * (GN / 16)) / 8, 256, 0, stream >>> (x, W, a, h, s2);
    gat_rowmax<<< GB,                   256, 0, stream >>> (s2, mmax);
    gat_buildg<<< GB * (GN / 128),      256, 0, stream >>> (h, s2, mmax, gt);
    gat_attn  <<< GB * (GN / 16),       256, 0, stream >>> (adj, gt, bias, out);
}
